// GaussianSpatialXFeatureLinear_6313601925730
// MI455X (gfx1250) — matrix operand order repaired and checked by reading, not compiled, not run on hardware
//
#include <hip/hip_runtime.h>

typedef _Float16 half16 __attribute__((ext_vector_type(16)));
typedef float   floatx8 __attribute__((ext_vector_type(8)));

#define CDIM   64
#define WH     1024          // 32*32 spatial positions
#define NOUT   4096
#define NBATCH 64
#define KCHUNK 128
#define LDSTR  136           // padded LDS row stride in halfs (128 + 8)

// ---------------------------------------------------------------------------
// CDNA5 async global->LDS copy (ASYNCcnt-tracked). The LDS aperture places
// the byte offset in addr[31:0] (ISA 10.2), so truncating the flat pointer
// yields the per-lane LDS address VGPR the instruction wants.
// ---------------------------------------------------------------------------
__device__ __forceinline__ void async_copy_b128(void* ldsPtr, const void* gPtr) {
    unsigned ldsOff = (unsigned)(unsigned long long)ldsPtr;
    asm volatile("global_load_async_to_lds_b128 %0, %1, off"
                 :: "v"(ldsOff), "v"(gPtr)
                 : "memory");
}
__device__ __forceinline__ void wait_async0() {
    asm volatile("s_wait_asynccnt 0" ::: "memory");
}

// ---------------------------------------------------------------------------
// Kernel A: S[n][p] = exp(-((cx-gx)^2 + (cy-gy)^2) / sigma^2), stored f16.
// p = ix*32 + iy, grid value = index - 15.5 (linspace(-15.5, 15.5, 32)).
// ---------------------------------------------------------------------------
__global__ void spatial_kernel(const float* __restrict__ cx,
                               const float* __restrict__ cy,
                               const float* __restrict__ sigma,
                               _Float16* __restrict__ S) {
    int n  = blockIdx.x;
    int p0 = threadIdx.x * 4;
    float cxv = cx[n], cyv = cy[n], sg = sigma[n];
    float inv = 1.0f / (sg * sg);
    union { _Float16 h[4]; uint2 u; } pk;
#pragma unroll
    for (int i = 0; i < 4; ++i) {
        int p = p0 + i;
        float gx = (float)(p >> 5) - 15.5f;
        float gy = (float)(p & 31) - 15.5f;
        float dx = cxv - gx, dy = cyv - gy;
        pk.h[i] = (_Float16)__expf(-(dx * dx + dy * dy) * inv);
    }
    *reinterpret_cast<uint2*>(S + (size_t)n * WH + p0) = pk.u;
}

// ---------------------------------------------------------------------------
// Kernel B: x fp32 -> f16 pre-pass (done once; the GEMM re-reads x 64 times,
// so converting up front halves A traffic and makes staging a pure copy).
// ---------------------------------------------------------------------------
__global__ void cvt_kernel(const float* __restrict__ x,
                           _Float16* __restrict__ Xh) {
    size_t i = ((size_t)blockIdx.x * blockDim.x + threadIdx.x) * 8;
    union { _Float16 h[8]; uint4 u; } pk;
#pragma unroll
    for (int j = 0; j < 8; j += 4) {
        float4 v = *reinterpret_cast<const float4*>(x + i + j);
        pk.h[j + 0] = (_Float16)v.x; pk.h[j + 1] = (_Float16)v.y;
        pk.h[j + 2] = (_Float16)v.z; pk.h[j + 3] = (_Float16)v.w;
    }
    *reinterpret_cast<uint4*>(Xh + i) = pk.u;
}

// ---------------------------------------------------------------------------
// Kernel C: fused GEMM + weighted channel reduction.
// Workgroup: 128 threads (4 wave32). Output tile: 16 batches x 64 neurons.
// Wave w owns neurons [ng*64 + w*16, +16) as one 16x16 WMMA C tile.
// y[N,n] = sum_{p-chunks} sum_c f[n,c] * WMMA(x_c_chunk, S_chunk) + b[n]
// A tile is double-buffered; all staging uses async global->LDS copies so
// the copy for channel c+1 overlaps the WMMAs of channel c (1 barrier / c).
// ---------------------------------------------------------------------------
__global__ __launch_bounds__(128)
void gemm_kernel(const _Float16* __restrict__ Xh,
                 const _Float16* __restrict__ S,
                 const float* __restrict__ feat,
                 const float* __restrict__ bias,
                 float* __restrict__ out) {
    __shared__ _Float16 ldsA[2][16 * LDSTR];  // x tile (double buffered)
    __shared__ _Float16 ldsB[64 * LDSTR];     // S tile: 64 neurons x 128 k
    __shared__ float    ldsF[CDIM * 64];      // features, [c][n_local]

    const int t    = threadIdx.x;
    const int lane = t & 31;
    const int wave = t >> 5;
    const int bt   = blockIdx.x;              // batch tile: rows bt*16 .. +15
    const int ng   = blockIdx.y;              // neuron group: cols ng*64 .. +63

    // Stage features into LDS transposed to [c][n_local] (bank-conflict free).
    for (int i = t; i < CDIM * 64; i += 128) {
        int nl = i >> 6, c = i & 63;
        ldsF[c * 64 + nl] = feat[(size_t)(ng * 64 + nl) * CDIM + c];
    }

    const int nl16 = lane & 15;
    const int hi   = lane >> 4;               // 0 for lanes 0-15, 1 for 16-31

    // Fragment bases (contiguous k = 16*hi + s, same order in A and B):
    //  A (16x32 f16): lane row M = nl16; hi-half lanes offset K by 16.
    //  B (32x16 f16): lane col = nl16;   hi-half lanes offset K by 16.
    const int aFragOff = nl16 * LDSTR + 16 * hi;
    const _Float16* bBase = &ldsB[(wave * 16 + nl16) * LDSTR + 16 * hi];

    // Per-thread staging coordinates.
    const int arow = t >> 3;                  // 0..15
    const int acol = (t & 7) * 16;            // 16 halfs per thread (2x b128)
    const int brow = t >> 1;                  // 0..63
    const int bseg = (t & 1) * 64;            // 64 halfs per thread (8x b128)

    const _Float16* aG0 =
        Xh + ((size_t)(bt * 16 + arow) * CDIM) * WH + acol;   // + c*WH + p0
    const _Float16* bG0 =
        S + (size_t)(ng * 64 + brow) * WH + bseg;             // + p0

    floatx8 accf = {};                        // final f32 accumulator (y tile)

    for (int p0 = 0; p0 < WH; p0 += KCHUNK) {
        // ---- issue async copies: B tile + A tile for c = 0 ----
        {
            const _Float16* g = bG0 + p0;
            _Float16*       l = &ldsB[brow * LDSTR + bseg];
#pragma unroll
            for (int i = 0; i < 8; ++i)
                async_copy_b128(l + i * 8, g + i * 8);
        }
        {
            const _Float16* g = aG0 + p0;     // c = 0
            _Float16*       l = &ldsA[0][arow * LDSTR + acol];
            async_copy_b128(l,     g);
            async_copy_b128(l + 8, g + 8);
        }
        wait_async0();
        __syncthreads();

        // Preload the 4 B fragments; reused across all 64 channels.
        half16 bfrag[4];
#pragma unroll
        for (int ks = 0; ks < 4; ++ks) {
            union { uint4 q[2]; half16 v; } u;
            const _Float16* src = bBase + ks * 32;
            u.q[0] = *reinterpret_cast<const uint4*>(src);
            u.q[1] = *reinterpret_cast<const uint4*>(src + 8);
            bfrag[ks] = u.v;
        }

        for (int c = 0; c < CDIM; ++c) {
            // Prefetch A tile for c+1 into the other buffer (async).
            if (c + 1 < CDIM) {
                const _Float16* g = aG0 + (size_t)(c + 1) * WH + p0;
                _Float16* l = &ldsA[(c + 1) & 1][arow * LDSTR + acol];
                async_copy_b128(l,     g);
                async_copy_b128(l + 8, g + 8);
            }

            // ---- 4 chained WMMAs over K = 128 for this channel ----
            const _Float16* aB = &ldsA[c & 1][aFragOff];
            floatx8 wacc = {};
#pragma unroll
            for (int ks = 0; ks < 4; ++ks) {
                union { uint4 q[2]; half16 v; } ua;
                const _Float16* src = aB + ks * 32;
                ua.q[0] = *reinterpret_cast<const uint4*>(src);
                ua.q[1] = *reinterpret_cast<const uint4*>(src + 8);
                wacc = __builtin_amdgcn_wmma_f32_16x16x32_f16(
                    false, ua.v, false, bfrag[ks], (short)0, wacc,
                    false, false);
            }

            // Weighted channel reduction: n is per-lane constant over VGPRs.
            float f = ldsF[c * 64 + wave * 16 + nl16];
#pragma unroll
            for (int r = 0; r < 8; ++r) accf[r] += f * wacc[r];

            wait_async0();                    // c+1 tile landed in LDS
            __syncthreads();                  // publish to all waves
        }
    }

    // D layout: lane<16 -> N=lane, M=r; lane>=16 -> N=lane-16, M=r+8.
    int ncol = ng * 64 + wave * 16 + nl16;
    float bv = bias[ncol];
#pragma unroll
    for (int r = 0; r < 8; ++r) {
        int M = r + 8 * hi;
        out[(size_t)(bt * 16 + M) * NOUT + ncol] = accf[r] + bv;
    }
}

// ---------------------------------------------------------------------------
extern "C" void kernel_launch(void* const* d_in, const int* in_sizes, int n_in,
                              void* d_out, int out_size, void* d_ws, size_t ws_size,
                              hipStream_t stream) {
    const float* x     = (const float*)d_in[0];   // [64,64,32,32]
    const float* cx    = (const float*)d_in[1];   // [4096]
    const float* cy    = (const float*)d_in[2];   // [4096]
    const float* sigma = (const float*)d_in[3];   // [4096]
    const float* feat  = (const float*)d_in[4];   // [4096,64]
    const float* bias  = (const float*)d_in[5];   // [4096]
    float* out = (float*)d_out;                   // [64,4096]

    _Float16* S  = (_Float16*)d_ws;                        // 8 MB
    _Float16* Xh = (_Float16*)d_ws + (size_t)NOUT * WH;    // 8 MB

    spatial_kernel<<<dim3(NOUT), dim3(256), 0, stream>>>(cx, cy, sigma, S);

    // x has 64*64*1024 = 4,194,304 elements; 8 per thread, 256 per block.
    cvt_kernel<<<dim3((NBATCH * CDIM * WH) / (256 * 8)), dim3(256), 0, stream>>>(
        x, Xh);

    gemm_kernel<<<dim3(NBATCH / 16, NOUT / 64), dim3(128), 0, stream>>>(
        Xh, S, feat, bias, out);
}
